// Crossgcn_block_38817914421573
// MI455X (gfx1250) — matrix operand order repaired and checked by reading, not compiled, not run on hardware
//
#include <hip/hip_runtime.h>
#include <hip/hip_bf16.h>

typedef unsigned short u16;
typedef unsigned int   u32;
typedef unsigned long long u64;
typedef __attribute__((ext_vector_type(4)))  u32    v4u;
typedef __attribute__((ext_vector_type(8)))  float  v8f;
typedef __attribute__((ext_vector_type(8)))  u32    v8u;
typedef __attribute__((ext_vector_type(16))) __bf16 v16bf;

#define BB 8
#define NN 1024
#define FF 16
#define TT 24
#define KK3 3
#define OO 64
#define CC 64
#define FT 384
#define BN 8192

static __device__ __forceinline__ u16 f2bf(float f) {
  u32 u = __float_as_uint(f);
  u32 r = (u + 0x7FFFu + ((u >> 16) & 1u)) >> 16;
  return (u16)r;
}

// ---------------- elementwise f32 -> bf16 ----------------
__global__ void k_f2bf(const float* __restrict__ in, u16* __restrict__ out, long n) {
  long i = (long)blockIdx.x * blockDim.x + threadIdx.x;
  if (i < n) out[i] = f2bf(in[i]);
}

// batched transpose-convert: in (R,C) row-major f32 -> out (C,R) bf16, batch = grid.z
__global__ void k_tr_bf(const float* __restrict__ in, u16* __restrict__ out, int R, int C) {
  long i = (long)blockIdx.x * blockDim.x + threadIdx.x;
  if (i >= (long)R * C) return;
  long bz = blockIdx.z;
  in  += bz * (long)R * C;
  out += bz * (long)R * C;
  int r = (int)(i / C), c = (int)(i % C);
  out[(long)c * R + r] = f2bf(in[i]);
}

// ---------------- WMMA bf16 GEMM ----------------
// C[M,N] = A(M,K,row-major) * B  with B supplied TRANSPOSED: Bt (N,K,row-major).
// Block = 128 threads (4 waves), 128x64 tile; wave w owns rows [tm+32w,+32) x 64 cols
// as 2 (M) x 4 (N) WMMA accumulators -> every B fragment feeds 2 WMMAs.
// K-tiles (BK=32) double-buffered in LDS via global_load_async_to_lds_b128 (ASYNCcnt).
// Fragments are pure ds_load_b128 (A row pairs; Bt rows are pair-packed columns).
// OUTMODE: 0 = f32 row-major, 1 = bf16 row-major, 2 = bf16 transposed (store C^T)
// grid = (N/64, M/128, batch)
template <int ACT, bool HAS_BCOL, bool HAS_BMAT, int OUTMODE>
__global__ void __launch_bounds__(128)
k_gemm(const u16* __restrict__ A, const u16* __restrict__ Bt,
       const float* __restrict__ biasCol, const float* __restrict__ biasMat,
       float* __restrict__ Cf, u16* __restrict__ Cbf,
       int lda, int ldb, int ldc, int Kd,
       long long sA, long long sB, long long sC, int bDivB) {
  __shared__ __attribute__((aligned(16))) u16 sAt[2][128 * 32]; // A tile: 128 rows x 32 k
  __shared__ __attribute__((aligned(16))) u16 sBt[2][64 * 32];  // Bt tile: 64 cols x 32 k
  const int tid = threadIdx.x;
  const int wave = tid >> 5, lane = tid & 31;
  const int ml = lane & 15, hf = lane >> 4;
  const int tn = blockIdx.x << 6, tm = blockIdx.y << 7;
  const int bz = blockIdx.z;
  A  += (long long)bz * sA;
  Bt += (long long)(bz / bDivB) * sB;
  v8f acc[2][4] = {};

  // Stage one 32-deep K tile (8KB A + 4KB Bt) into buffer pb. 16B per async issue.
  auto stage = [&](int k0, int pb) {
    const u32 lA = (u32)(uintptr_t)&sAt[pb][0];  // LDS aperture: low 32 bits = offset
    const u32 lB = (u32)(uintptr_t)&sBt[pb][0];
#pragma unroll
    for (int it = 0; it < 4; it++) {             // A: 128 rows x 4 16B segments
      int c = tid + (it << 7);
      int row = c >> 2, seg = c & 3;
      const u16* g = A + (long long)(tm + row) * lda + k0 + (seg << 3);
      u32 l = lA + (((row << 5) + (seg << 3)) << 1);
      asm volatile("global_load_async_to_lds_b128 %0, %1, off"
                   :: "v"(l), "v"((u64)(uintptr_t)g) : "memory");
    }
#pragma unroll
    for (int it = 0; it < 2; it++) {             // Bt: 64 cols x 4 16B segments
      int c = tid + (it << 7);
      int row = c >> 2, seg = c & 3;
      const u16* g = Bt + (long long)(tn + row) * ldb + k0 + (seg << 3);
      u32 l = lB + (((row << 5) + (seg << 3)) << 1);
      asm volatile("global_load_async_to_lds_b128 %0, %1, off"
                   :: "v"(l), "v"((u64)(uintptr_t)g) : "memory");
    }
  };

  const int nkt = Kd >> 5;
  stage(0, 0);
  int p = 0;
  for (int ki = 0; ki < nkt; ki++) {
    if (ki + 1 < nkt) {
      stage((ki + 1) << 5, p ^ 1);                     // overlap DMA with compute
      asm volatile("s_wait_asynccnt 6" ::: "memory");  // in-order: tile ki has landed
    } else {
      asm volatile("s_wait_asynccnt 0" ::: "memory");
    }
    __syncthreads();

    // A fragments (ISA 7.12.2 16-bit A 16x32): two 4-dword runs per lane (k 8h.., 16+8h..) -> 2x b128
    v16bf av[2];
#pragma unroll
    for (int g = 0; g < 2; g++) {
      const u16* ap = &sAt[p][((wave << 5) + (g << 4) + ml) << 5];
      v4u a0 = *(const v4u*)&ap[hf << 3];
      v4u a1 = *(const v4u*)&ap[16 + (hf << 3)];
      av[g] = __builtin_bit_cast(v16bf, __builtin_shufflevector(a0, a1, 0, 1, 2, 3, 4, 5, 6, 7));
    }

#pragma unroll
    for (int s = 0; s < 4; s++) {
      // B fragment: Bt row (= output col) holds pair-packed column -> 2x b128
      const u16* bp = &sBt[p][((s << 4) + ml) << 5];
      v4u b0 = *(const v4u*)&bp[hf << 3];
      v4u b1 = *(const v4u*)&bp[16 + (hf << 3)];
      v16bf bv = __builtin_bit_cast(v16bf, __builtin_shufflevector(b0, b1, 0, 1, 2, 3, 4, 5, 6, 7));
#pragma unroll
      for (int g = 0; g < 2; g++)
        acc[g][s] = __builtin_amdgcn_wmma_f32_16x16x32_bf16(
            false, av[g], false, bv, (short)0, acc[g][s], false, false);
    }
    __syncthreads();
    p ^= 1;
  }

  // ---- epilogue (compile-time specialized) ----
#pragma unroll
  for (int g = 0; g < 2; g++) {
#pragma unroll
    for (int s = 0; s < 4; s++) {
      v8f a = acc[g][s];
      const int gc = tn + (s << 4) + ml;
      const float bc = HAS_BCOL ? biasCol[gc] : 0.f;
#pragma unroll
      for (int r = 0; r < 8; r++) {
        int gr = tm + (wave << 5) + (g << 4) + (hf << 3) + r; // f32 C/D: vgpr r -> row (+8 hi)
        float v = a[r] + bc;
        if (HAS_BMAT) v += biasMat[(long long)gr * ldc + gc];
        if (ACT == 2) v = 1.f / (1.f + __expf(-v));
        if (OUTMODE == 0) Cf [(long long)bz * sC + (long long)gr * ldc + gc] = v;
        if (OUTMODE == 1) Cbf[(long long)bz * sC + (long long)gr * ldc + gc] = f2bf(v);
        if (OUTMODE == 2) Cbf[(long long)bz * sC + (long long)gc * ldc + gr] = f2bf(v);
      }
    }
  }
}

// ---------------- temporal attention: one wave per (b,n) ----------------
__global__ void k_tat(const float* __restrict__ h, const float* __restrict__ inw,
                      const float* __restrict__ inb, const float* __restrict__ outw,
                      const float* __restrict__ outb, float* __restrict__ t_at) {
  __shared__ float kv[4][2][16][24];
  const int wave = threadIdx.x >> 5, lane = threadIdx.x & 31;
  const long bn = (long)blockIdx.x * 4 + wave;
  float hrow[24], q[24];
  if (lane < 16) {
    const float* hp = h + bn * 384 + lane * 24;
#pragma unroll
    for (int t = 0; t < 24; t++) hrow[t] = hp[t];
    for (int tp = 0; tp < 24; tp++) {
      float aq = inb[tp], ak = inb[24 + tp], av = inb[48 + tp];
#pragma unroll
      for (int t = 0; t < 24; t++) {
        aq += hrow[t] * inw[tp * 24 + t];
        ak += hrow[t] * inw[(24 + tp) * 24 + t];
        av += hrow[t] * inw[(48 + tp) * 24 + t];
      }
      q[tp] = aq; kv[wave][0][lane][tp] = ak; kv[wave][1][lane][tp] = av;
    }
  }
  __syncthreads();
  if (lane < 16) {
    const float scale = 0.20412414523193154f; // 1/sqrt(24)
    float s[16], mx = -1e30f;
    for (int j = 0; j < 16; j++) {
      float a = 0.f;
#pragma unroll
      for (int t = 0; t < 24; t++) a += q[t] * kv[wave][0][j][t];
      s[j] = a * scale; mx = fmaxf(mx, s[j]);
    }
    float se = 0.f;
    for (int j = 0; j < 16; j++) { s[j] = __expf(s[j] - mx); se += s[j]; }
    const float inv = 1.f / se;
    float o[24];
    for (int t = 0; t < 24; t++) {
      float a = 0.f;
#pragma unroll
      for (int j = 0; j < 16; j++) a += s[j] * kv[wave][1][j][t];
      o[t] = a * inv;
    }
    float* op = t_at + bn * 384 + lane * 24;
    for (int tp = 0; tp < 24; tp++) {
      float a = outb[tp];
#pragma unroll
      for (int t = 0; t < 24; t++) a += o[t] * outw[tp * 24 + t];
      op[tp] = a;
    }
  }
}

// tas[b, f*T+t, n] = bf16(t_at[b,n,f,t])
__global__ void k_build_tas(const float* __restrict__ t_at, u16* __restrict__ tas) {
  long i = (long)blockIdx.x * blockDim.x + threadIdx.x;
  if (i >= (long)BN * 384) return;
  int t = (int)(i % 24); int f = (int)((i / 24) % 16);
  int n = (int)((i / 384) % NN); int b = (int)(i / ((long)NN * 384));
  tas[((long)b * 384 + f * 24 + t) * NN + n] = f2bf(t_at[i]);
}

// sc[b,h,f] = sum_d qr[b,h,d]*router[f,h,d]; softmax over f -> aw
__global__ void k_sc(const float* __restrict__ qr, const float* __restrict__ router,
                     float* __restrict__ aw) {
  const int bh = blockIdx.x, h = bh % FT, tid = threadIdx.x;
  float part[24];
#pragma unroll
  for (int f = 0; f < 24; f++) part[f] = 0.f;
  const float* qp = qr + (long)bh * NN;
  for (int d = tid; d < NN; d += 256) {
    float qv = qp[d];
#pragma unroll
    for (int f = 0; f < 24; f++) part[f] += qv * router[((long)f * FT + h) * NN + d];
  }
  __shared__ float buf[256]; __shared__ float sf[24];
  for (int f = 0; f < 24; f++) {
    buf[tid] = part[f]; __syncthreads();
    for (int s = 128; s > 0; s >>= 1) { if (tid < s) buf[tid] += buf[tid + s]; __syncthreads(); }
    if (tid == 0) sf[f] = buf[0];
    __syncthreads();
  }
  if (tid == 0) {
    float mx = -1e30f;
    for (int f = 0; f < 24; f++) mx = fmaxf(mx, sf[f]);
    float e[24], se = 0.f;
    for (int f = 0; f < 24; f++) { e[f] = __expf(sf[f] - mx); se += e[f]; }
    float inv = 1.f / se;
    for (int f = 0; f < 24; f++) aw[(long)bh * 24 + f] = e[f] * inv;
  }
}

// recv[b,h,d] = sum_f aw[b,h,f]*router[f,h,d]  -> bf16
__global__ void k_recv(const float* __restrict__ aw, const float* __restrict__ router,
                       u16* __restrict__ recv) {
  long i = (long)blockIdx.x * blockDim.x + threadIdx.x;
  if (i >= (long)BB * FT * NN) return;
  int d = (int)(i & (NN - 1)); long bh = i >> 10; int h = (int)(bh % FT);
  float s = 0.f;
#pragma unroll
  for (int f = 0; f < 24; f++) s += aw[bh * 24 + f] * router[((long)f * FT + h) * NN + d];
  recv[i] = f2bf(s);
}

// From raw (B,FT,N): lhs (M,K=32 zero-padded) and rhs in Bt layout (N rows, K=32)
__global__ void k_lhsrhs(const float* __restrict__ raw, const float* __restrict__ W1,
                         const float* __restrict__ W2, const float* __restrict__ W3,
                         u16* __restrict__ lhsbf, u16* __restrict__ rhsBt) {
  int bn = blockIdx.x * blockDim.x + threadIdx.x;
  if (bn >= BN) return;
  int b = bn >> 10, n = bn & (NN - 1);
  const float* rp = raw + (long)b * FT * NN + n;
  float l0[16], rh[24];
#pragma unroll
  for (int t = 0; t < 24; t++) rh[t] = 0.f;
  for (int f = 0; f < 16; f++) {
    float a = 0.f, w3 = W3[f];
#pragma unroll
    for (int t = 0; t < 24; t++) {
      float v = rp[(long)(f * 24 + t) * NN];
      a += v * W1[t]; rh[t] += w3 * v;
    }
    l0[f] = a;
  }
  for (int tp = 0; tp < 32; tp++) {
    float a = 0.f;
    if (tp < 24) {
#pragma unroll
      for (int f = 0; f < 16; f++) a += l0[f] * W2[f * 24 + tp];
    }
    lhsbf[(long)bn * 32 + tp]  = f2bf(tp < 24 ? a : 0.f);
    rhsBt[(long)bn * 32 + tp] = f2bf(tp < 24 ? rh[tp] : 0.f);  // Bt row = output col m
  }
}

// softmax over axis=1 (rows i) of S[b,i,j], in place; one block per (b,j)
__global__ void k_softmax_col(float* __restrict__ S) {
  const int bj = blockIdx.x, b = bj >> 10, j = bj & (NN - 1), tid = threadIdx.x;
  float* col = S + (long)b * NN * NN + j;
  __shared__ float buf[256];
  float mx = -1e30f;
  for (int i = tid; i < NN; i += 256) mx = fmaxf(mx, col[(long)i * NN]);
  buf[tid] = mx; __syncthreads();
  for (int s = 128; s > 0; s >>= 1) { if (tid < s) buf[tid] = fmaxf(buf[tid], buf[tid + s]); __syncthreads(); }
  mx = buf[0]; __syncthreads();
  float se = 0.f;
  for (int i = tid; i < NN; i += 256) se += __expf(col[(long)i * NN] - mx);
  buf[tid] = se; __syncthreads();
  for (int s = 128; s > 0; s >>= 1) { if (tid < s) buf[tid] += buf[tid + s]; __syncthreads(); }
  float inv = 1.f / buf[0];
  for (int i = tid; i < NN; i += 256) { long o = (long)i * NN; col[o] = __expf(col[o] - mx) * inv; }
}

// TSt[b,k,m,n] = bf16( Tk[k,n,m] * S[b,n,m] )  (pre-transposed A for cheb GEMM)
__global__ void k_tst(const float* __restrict__ Tk, const float* __restrict__ S,
                      u16* __restrict__ TSt) {
  int n = blockIdx.x * 256 + threadIdx.x;
  int m = blockIdx.y, bk = blockIdx.z, b = bk / 3, k = bk % 3;
  TSt[((long)bk * NN + m) * NN + n] =
      f2bf(Tk[((long)k * NN + n) * NN + m] * S[((long)b * NN + n) * NN + m]);
}

// sg[b,m,o,t] = relu( sum_{k,f} r[b,k,m,f*T+t] * Theta[k,f,o] ); block per (b,m)
__global__ void k_sg(const float* __restrict__ r, const float* __restrict__ Theta,
                     float* __restrict__ sg) {
  __shared__ float rs[3 * 384];
  __shared__ float th[3 * 16 * 64];
  const long bm = blockIdx.x; const int b = (int)(bm >> 10), m = (int)(bm & (NN - 1));
  const int tid = threadIdx.x;
  for (int i = tid; i < 1152; i += 256)
    rs[i] = r[((long)(b * 3 + i / 384) * NN + m) * 384 + (i % 384)];
  for (int i = tid; i < 3072; i += 256) th[i] = Theta[i];
  __syncthreads();
  for (int idx = tid; idx < OO * TT; idx += 256) {
    int t = idx % 24, o = idx / 24;
    float acc = 0.f;
#pragma unroll
    for (int k = 0; k < 3; k++)
#pragma unroll
      for (int f = 0; f < 16; f++)
        acc += rs[k * 384 + f * 24 + t] * th[(k * 16 + f) * 64 + o];
    sg[(bm * OO + o) * TT + t] = fmaxf(acc, 0.f);
  }
}

// time conv + residual 1x1 conv + relu + layernorm(C) -> out (B,N,C,T); block(64) per (b,n)
__global__ void k_final(const float* __restrict__ sg, const float* __restrict__ x,
                        const float* __restrict__ tcw, const float* __restrict__ tcb,
                        const float* __restrict__ rcw, const float* __restrict__ rcb,
                        const float* __restrict__ lng, const float* __restrict__ lnb,
                        float* __restrict__ out) {
  __shared__ float sgt[OO * TT];
  __shared__ float xt[FF * TT];
  __shared__ float zt[TT * CC];
  __shared__ float mvs[TT * 2];
  const long bn = blockIdx.x; const int c = threadIdx.x;
  for (int i = c; i < OO * TT; i += 64) sgt[i] = sg[bn * OO * TT + i];
  for (int i = c; i < FF * TT; i += 64) xt[i] = x[bn * 384 + i];
  __syncthreads();
  for (int t = 0; t < TT; t++) {
    float acc = tcb[c];
#pragma unroll
    for (int dt = 0; dt < 3; dt++) {
      int tt = t + dt - 1;
      if (tt >= 0 && tt < TT) {
#pragma unroll
        for (int o = 0; o < OO; o++) acc += sgt[o * TT + tt] * tcw[(c * OO + o) * 3 + dt];
      }
    }
    float res = rcb[c];
#pragma unroll
    for (int f = 0; f < FF; f++) res += rcw[c * FF + f] * xt[f * TT + t];
    zt[t * CC + c] = fmaxf(acc + res, 0.f);
  }
  __syncthreads();
  if (c < TT) {
    float m = 0.f;
    for (int j = 0; j < CC; j++) m += zt[c * CC + j];
    m *= (1.f / CC);
    float v = 0.f;
    for (int j = 0; j < CC; j++) { float d = zt[c * CC + j] - m; v += d * d; }
    v *= (1.f / CC);
    mvs[c] = m; mvs[TT + c] = __frsqrt_rn(v + 1e-5f);
  }
  __syncthreads();
  const float g = lng[c], be = lnb[c];
  for (int t = 0; t < TT; t++) {
    float val = (zt[t * CC + c] - mvs[t]) * mvs[TT + t] * g + be;
    out[(bn * CC + c) * TT + t] = val;
  }
}

extern "C" void kernel_launch(void* const* d_in, const int* in_sizes, int n_in,
                              void* d_out, int out_size, void* d_ws, size_t ws_size,
                              hipStream_t stream) {
  const float* x       = (const float*)d_in[0];
  const float* l1w     = (const float*)d_in[1];
  const float* l1b     = (const float*)d_in[2];
  const float* l2w     = (const float*)d_in[3];
  const float* l2b     = (const float*)d_in[4];
  const float* inw     = (const float*)d_in[5];
  const float* inb     = (const float*)d_in[6];
  const float* outw    = (const float*)d_in[7];
  const float* outb    = (const float*)d_in[8];
  const float* W1      = (const float*)d_in[9];
  const float* W2      = (const float*)d_in[10];
  const float* W3      = (const float*)d_in[11];
  const float* bs      = (const float*)d_in[12];
  const float* Vs      = (const float*)d_in[13];
  const float* router  = (const float*)d_in[14];
  const float* qw      = (const float*)d_in[15];
  const float* qb      = (const float*)d_in[16];
  const float* ow      = (const float*)d_in[17];
  const float* ob      = (const float*)d_in[18];
  const float* Tk      = (const float*)d_in[19];
  const float* Theta   = (const float*)d_in[20];
  const float* tcw     = (const float*)d_in[21];
  const float* tcb     = (const float*)d_in[22];
  const float* rcw     = (const float*)d_in[23];
  const float* rcb     = (const float*)d_in[24];
  const float* lng     = (const float*)d_in[25];
  const float* lnb     = (const float*)d_in[26];
  float* out = (float*)d_out;

  char* ws = (char*)d_ws;
  size_t off = 0;
  auto alloc = [&](size_t bytes) -> char* {
    off = (off + 255) & ~(size_t)255;
    char* p = ws + off; off += bytes; return p;
  };
  u16*   xbf    = (u16*)  alloc((size_t)BN * 384 * 2);           // A for GEMM1
  u16*   xbt    = (u16*)  alloc((size_t)BN * 384 * 2);           // Bt for cheb (B,384,N)
  u16*   w1bf   = (u16*)  alloc((size_t)512 * 384 * 2);          // Bt == l1w layout
  u16*   w2bf   = (u16*)  alloc((size_t)384 * 512 * 2);          // Bt == l2w layout
  u16*   qwbf   = (u16*)  alloc((size_t)NN * NN * 2);            // Bt == ra_q_w layout
  u16*   owbf   = (u16*)  alloc((size_t)NN * NN * 2);            // Bt == ra_o_w layout
  u16*   Vsbf   = (u16*)  alloc((size_t)NN * NN * 2);
  float* aw     = (float*)alloc((size_t)BB * FT * 24 * 4);
  u16*   lhsbf  = (u16*)  alloc((size_t)BN * 32 * 2);
  u16*   rhsBt  = (u16*)  alloc((size_t)BN * 32 * 2);
  char*  slotA  =         alloc((size_t)BN * 384 * 4);           // h2 -> qr -> raw
  char*  slotB  =         alloc((size_t)BN * 384 * 4);           // t_at
  char*  slotC  =         alloc((size_t)BN * 512 * 2);           // h1bf -> tasbf -> recvbf
  u16*   sigT   = (u16*)  alloc((size_t)BB * NN * NN * 2);       // sigmoid(prod)^T, Bt for S
  float* S      = (float*)alloc((size_t)BB * NN * NN * 4);
  char*  slotD  =         alloc((size_t)BB * KK3 * NN * NN * 2); // TSt -> reused by sg
  float* rbuf   = (float*)alloc((size_t)BB * KK3 * NN * 384 * 4);

  float* h2     = (float*)slotA;
  float* qr     = (float*)slotA;
  float* raw    = (float*)slotA;
  float* t_at   = (float*)slotB;
  u16*   h1bf   = (u16*)  slotC;
  u16*   tasbf  = (u16*)  slotC;
  u16*   recvbf = (u16*)  slotC;
  u16*   TSt    = (u16*)  slotD;
  float* sg     = (float*)slotD;

  // -- precision staging (all Bt operands are the weights' native layout) --
  k_f2bf<<<(BN * 384 + 255) / 256, 256, 0, stream>>>(x, xbf, (long)BN * 384);
  k_tr_bf<<<dim3((NN * 384 + 255) / 256, 1, BB), 256, 0, stream>>>(x, xbt, NN, 384);
  k_f2bf<<<(512 * 384 + 255) / 256, 256, 0, stream>>>(l1w, w1bf, 512 * 384);
  k_f2bf<<<(384 * 512 + 255) / 256, 256, 0, stream>>>(l2w, w2bf, 384 * 512);
  k_f2bf<<<(NN * NN + 255) / 256, 256, 0, stream>>>(qw, qwbf, (long)NN * NN);
  k_f2bf<<<(NN * NN + 255) / 256, 256, 0, stream>>>(ow, owbf, (long)NN * NN);
  k_f2bf<<<(NN * NN + 255) / 256, 256, 0, stream>>>(Vs, Vsbf, (long)NN * NN);

  // -- temporal MLP (WMMA) ------------------------------------------------
  k_gemm<0, true, false, 1><<<dim3(512 / 64, BN / 128, 1), 128, 0, stream>>>(
      xbf, w1bf, l1b, nullptr, nullptr, h1bf, 384, 384, 512, 384, 0, 0, 0, 1);
  k_gemm<0, true, false, 0><<<dim3(384 / 64, BN / 128, 1), 128, 0, stream>>>(
      h1bf, w2bf, l2b, nullptr, h2, nullptr, 512, 512, 384, 512, 0, 0, 0, 1);

  // -- temporal self-attention over T=24 ----------------------------------
  k_tat<<<BN / 4, 128, 0, stream>>>(h2, inw, inb, outw, outb, t_at);
  k_build_tas<<<(BN * 384 + 255) / 256, 256, 0, stream>>>(t_at, tasbf);

  // -- router attention (WMMA projections) --------------------------------
  k_gemm<0, true, false, 0><<<dim3(NN / 64, (BB * FT) / 128, 1), 128, 0, stream>>>(
      tasbf, qwbf, qb, nullptr, qr, nullptr, NN, NN, NN, NN, 0, 0, 0, 1);
  k_sc<<<BB * FT, 256, 0, stream>>>(qr, router, aw);
  k_recv<<<(BB * FT * NN + 255) / 256, 256, 0, stream>>>(aw, router, recvbf);
  k_gemm<0, true, false, 0><<<dim3(NN / 64, (BB * FT) / 128, 1), 128, 0, stream>>>(
      recvbf, owbf, ob, nullptr, raw, nullptr, NN, NN, NN, NN, 0, 0, 0, 1);

  // -- spatial attention ---------------------------------------------------
  k_lhsrhs<<<BN / 256, 256, 0, stream>>>(raw, W1, W2, W3, lhsbf, rhsBt);
  // sigT = sigmoid(lhs @ rhs^T + bs)^T  (transposed bf16 epilogue -> Bt for S GEMM)
  k_gemm<2, false, true, 2><<<dim3(NN / 64, NN / 128, BB), 128, 0, stream>>>(
      lhsbf, rhsBt, nullptr, bs, nullptr, sigT, 32, 32, NN, 32,
      (long long)NN * 32, (long long)NN * 32, (long long)NN * NN, 1);
  // S = Vs @ sig  (A shared across batch: sA = 0; B supplied transposed = sigT)
  k_gemm<0, false, false, 0><<<dim3(NN / 64, NN / 128, BB), 128, 0, stream>>>(
      Vsbf, sigT, nullptr, nullptr, S, nullptr, NN, NN, NN, NN,
      0, (long long)NN * NN, (long long)NN * NN, 1);
  k_softmax_col<<<BB * NN, 256, 0, stream>>>(S);

  // -- Chebyshev graph conv (WMMA batched) ---------------------------------
  k_tst<<<dim3(NN / 256, NN, BB * KK3), 256, 0, stream>>>(Tk, S, TSt);
  k_gemm<0, false, false, 0><<<dim3(384 / 64, NN / 128, BB * KK3), 128, 0, stream>>>(
      TSt, xbt, nullptr, nullptr, rbuf, nullptr, NN, NN, 384, NN,
      (long long)NN * NN, (long long)384 * NN, (long long)NN * 384, KK3);
  k_sg<<<BN, 256, 0, stream>>>(rbuf, Theta, sg);

  // -- time conv + residual + relu + layernorm -----------------------------
  k_final<<<BN, 64, 0, stream>>>(sg, x, tcw, tcb, rcw, rcb, lng, lnb, out);
}
